// TextHeading_84035330113648
// MI455X (gfx1250) — matrix operand order repaired and checked by reading, not compiled, not run on hardware
//
#include <hip/hip_runtime.h>

typedef __attribute__((ext_vector_type(16))) __bf16 v16bf;
typedef __attribute__((ext_vector_type(8)))  float  v8f;
typedef int v4i __attribute__((vector_size(4 * sizeof(int))));
typedef __bf16 bf16;

#define LDS_ROW 80   // 64B of tile data + 16B pad per row -> conflict-free b128 loads

#if __has_builtin(__builtin_amdgcn_global_load_async_to_lds_b128) && \
    __has_builtin(__builtin_amdgcn_s_wait_asynccnt)
#define USE_ASYNC_LDS 1
#else
#define USE_ASYNC_LDS 0
#endif

// builtin wants (v4i32 addrspace(1)*, lds ptr, i32 offset, i32 cpol)
#define AS1P(p) ((__attribute__((address_space(1))) v4i*)(p))
#define AS3P(p) ((__attribute__((address_space(3))) v4i*)(p))

union Frag { uint4 u[2]; v16bf v; };

// ---------------------------------------------------------------------------
// Conversion kernels
// ---------------------------------------------------------------------------
__global__ __launch_bounds__(256) void cvt_flat(const float* __restrict__ in,
                                                bf16* __restrict__ out, int n) {
    int i = blockIdx.x * 256 + threadIdx.x;
    if (i < n) out[i] = (bf16)in[i];
}

// words (128,128,768) f32 -> padded (128,132,768) bf16, pad rows zeroed
__global__ __launch_bounds__(256) void cvt_words(const float* __restrict__ in,
                                                 bf16* __restrict__ out, int n) {
    int i = blockIdx.x * 256 + threadIdx.x;
    if (i >= n) return;
    int d = i % 768;
    int t = i / 768;
    int lp = t % 132;
    int b  = t / 132;
    float v = (lp < 128) ? in[(b * 128 + lp) * 768 + d] : 0.0f;
    out[i] = (bf16)v;
}

// ---------------------------------------------------------------------------
// Fused multi-branch WMMA GEMM.
//   D[m,n] over branches j: acc_j = sum_k A[rowbase(m)+k] * Wj[n*Kj + k]
//   code  = positional max over relu(acc_j + bias_j)  (RELU=true path)
// Block tile 128x128, 8 waves (4M x 2N), wave tile 32x64 (2x4 subtiles).
// A row m maps to (b = m/RPB, l = m%RPB), row base = b*batchPitch + l*768.
// Staging: GLOBAL_LOAD_ASYNC_TO_LDS_B128 double-buffered, ASYNCcnt-paced.
// ---------------------------------------------------------------------------
template <int NBR, bool RELU>
__global__ __launch_bounds__(256, 1)
void wmma_gemm(const bf16* __restrict__ Abase, int RPB, int batchPitch,
               const bf16* __restrict__ Wa, int Ka, const float* __restrict__ ba, int la,
               const bf16* __restrict__ Wb, int Kb, const float* __restrict__ bbp, int lb,
               const bf16* __restrict__ Wc, int Kc, const float* __restrict__ bc, int lc,
               float* __restrict__ out) {
    __shared__ char ldsA[2][128 * LDS_ROW];
    __shared__ char ldsB[2][128 * LDS_ROW];

    const int tid   = threadIdx.x;
    const int lane  = tid & 31;
    const int l15   = lane & 15;
    const int khalf = lane >> 4;
    const int wave  = tid >> 5;

    const int bm = blockIdx.y;            // M block (128 rows)
    const int bn = blockIdx.x;            // N block (128 cols)
    const int waveM = (wave >> 1) * 32;   // 0,32,64,96
    const int waveN = (wave & 1) * 64;    // 0,64

    // staging: thread -> (row, 32B half) of the 128x32 bf16 tile
    const int sr = tid >> 1;
    const int sh = tid & 1;
    const int m  = bm * 128 + sr;
    const int rowA = (m / RPB) * batchPitch + (m % RPB) * 768;  // element offset
    const int nrow = bn * 128 + sr;
    const int ldsOff = sr * LDS_ROW + sh * 32;

    const bf16*  Wbr[3]  = {Wa, Wb, Wc};
    const int    Kbr[3]  = {Ka, Kb, Kc};
    const float* Bbr[3]  = {ba, bbp, bc};
    const int    Lim[3]  = {la, lb, lc};

    v8f code[2][4];

    // per-element position index l for the branch-validity masks
    int lrow[2][8];
#pragma unroll
    for (int ms = 0; ms < 2; ++ms)
#pragma unroll
        for (int e = 0; e < 8; ++e)
            lrow[ms][e] = (bm * 128 + waveM + ms * 16 + khalf * 8 + e) % RPB;

#pragma unroll
    for (int br = 0; br < NBR; ++br) {
        const bf16* W    = Wbr[br];
        const int   Klen = Kbr[br];
        const int   rowB = nrow * Klen;
        const int   nk   = Klen >> 5;

        v8f acc[2][4];
#pragma unroll
        for (int ms = 0; ms < 2; ++ms)
#pragma unroll
            for (int ns = 0; ns < 4; ++ns)
                acc[ms][ns] = (v8f){0.f, 0.f, 0.f, 0.f, 0.f, 0.f, 0.f, 0.f};

#if USE_ASYNC_LDS
        // Issue one 128x32 bf16 tile of A and B into lds buffer `buf`:
        // 4 async b128 DMAs per thread, tracked on ASYNCcnt (no VGPR staging).
        auto issue_tile = [&](int buf, int ko) {
            const bf16* gA = Abase + rowA + ko + sh * 16;
            const bf16* gB = W + rowB + ko + sh * 16;
            __builtin_amdgcn_global_load_async_to_lds_b128(
                AS1P(gA),     AS3P(&ldsA[buf][ldsOff]),      0, 0);
            __builtin_amdgcn_global_load_async_to_lds_b128(
                AS1P(gA + 8), AS3P(&ldsA[buf][ldsOff + 16]), 0, 0);
            __builtin_amdgcn_global_load_async_to_lds_b128(
                AS1P(gB),     AS3P(&ldsB[buf][ldsOff]),      0, 0);
            __builtin_amdgcn_global_load_async_to_lds_b128(
                AS1P(gB + 8), AS3P(&ldsB[buf][ldsOff + 16]), 0, 0);
        };
        issue_tile(0, 0);
#else
        {   // fallback: register staging
            const uint4* ga = (const uint4*)(Abase + rowA + sh * 16);
            const uint4* gb = (const uint4*)(W + rowB + sh * 16);
            uint4 a0 = ga[0], a1 = ga[1];
            uint4 b0 = gb[0], b1 = gb[1];
            *(uint4*)(&ldsA[0][ldsOff])      = a0;
            *(uint4*)(&ldsA[0][ldsOff + 16]) = a1;
            *(uint4*)(&ldsB[0][ldsOff])      = b0;
            *(uint4*)(&ldsB[0][ldsOff + 16]) = b1;
            __syncthreads();
        }
#endif

        for (int ks = 0; ks < nk; ++ks) {
            const int cur  = ks & 1;
            const int nxt  = cur ^ 1;
            const bool more = (ks + 1) < nk;

#if USE_ASYNC_LDS
            // prefetch next tile into the other buffer (overlaps WMMA below)
            if (more) issue_tile(nxt, (ks + 1) * 32);
            // tile `cur` (issued last iteration) must be resident; allow the
            // 4 just-issued DMAs for tile `nxt` to stay in flight
            if (more) __builtin_amdgcn_s_wait_asynccnt(4);
            else      __builtin_amdgcn_s_wait_asynccnt(0);
            __syncthreads();     // cross-wave: everyone's tile-cur DMAs landed
#else
            uint4 na0, na1, nb0, nb1;
            if (more) {
                const int ko = (ks + 1) * 32 + sh * 16;
                const uint4* ga = (const uint4*)(Abase + rowA + ko);
                const uint4* gb = (const uint4*)(W + rowB + ko);
                na0 = ga[0]; na1 = ga[1];
                nb0 = gb[0]; nb1 = gb[1];
            }
#endif

            // A frags: 16x32 bf16 layout (lane=M, khalf selects K halves)
            Frag fa[2], fb[4];
#pragma unroll
            for (int ms = 0; ms < 2; ++ms) {
                const int r = waveM + ms * 16 + l15;
                fa[ms].u[0] = *(const uint4*)(&ldsA[cur][r * LDS_ROW + khalf * 16]);
                fa[ms].u[1] = *(const uint4*)(&ldsA[cur][r * LDS_ROW + 32 + khalf * 16]);
            }
            // B frags: stored N-major; lane=N holds K 8*khalf..+7 and 16+8*khalf..+7
#pragma unroll
            for (int ns = 0; ns < 4; ++ns) {
                const int r = waveN + ns * 16 + l15;
                fb[ns].u[0] = *(const uint4*)(&ldsB[cur][r * LDS_ROW + khalf * 16]);
                fb[ns].u[1] = *(const uint4*)(&ldsB[cur][r * LDS_ROW + 32 + khalf * 16]);
            }

#pragma unroll
            for (int ms = 0; ms < 2; ++ms)
#pragma unroll
                for (int ns = 0; ns < 4; ++ns)
                    acc[ms][ns] = __builtin_amdgcn_wmma_f32_16x16x32_bf16(
                        false, fa[ms].v, false, fb[ns].v,
                        (short)0, acc[ms][ns], false, false);

            __syncthreads();  // all waves done reading lds[cur] before it is refilled
#if !USE_ASYNC_LDS
            if (more) {
                *(uint4*)(&ldsA[nxt][ldsOff])      = na0;
                *(uint4*)(&ldsA[nxt][ldsOff + 16]) = na1;
                *(uint4*)(&ldsB[nxt][ldsOff])      = nb0;
                *(uint4*)(&ldsB[nxt][ldsOff + 16]) = nb1;
                __syncthreads();
            }
#endif
        }

        // branch epilogue: bias (+relu) then fold into positional max
        const float* bias = Bbr[br];
        const int    lim  = Lim[br];
#pragma unroll
        for (int ms = 0; ms < 2; ++ms)
#pragma unroll
            for (int ns = 0; ns < 4; ++ns) {
                const float bv = bias[bn * 128 + waveN + ns * 16 + l15];
#pragma unroll
                for (int e = 0; e < 8; ++e) {
                    float v = acc[ms][ns][e] + bv;
                    if (RELU) v = fmaxf(v, 0.0f);
                    if (br == 0)
                        code[ms][ns][e] = v;
                    else
                        code[ms][ns][e] = (lrow[ms][e] < lim)
                                              ? fmaxf(code[ms][ns][e], v)
                                              : code[ms][ns][e];
                }
            }
    }

    // store merged result (pre-normalization) to workspace, 64B coalesced runs
#pragma unroll
    for (int ms = 0; ms < 2; ++ms)
#pragma unroll
        for (int ns = 0; ns < 4; ++ns)
#pragma unroll
            for (int e = 0; e < 8; ++e) {
                const int mg = bm * 128 + waveM + ms * 16 + khalf * 8 + e;
                const int ng = bn * 128 + waveN + ns * 16 + l15;
                out[mg * 1024 + ng] = code[ms][ns][e];
            }
}

// ---------------------------------------------------------------------------
// Sentence head: per-row L2 normalize (128 rows x 1024)
// ---------------------------------------------------------------------------
__global__ __launch_bounds__(256) void sent_norm(const float* __restrict__ raw,
                                                 float* __restrict__ out) {
    __shared__ float red[8];
    __shared__ float invs;
    const int row = blockIdx.x;
    const float* r = raw + row * 1024;
    float s = 0.0f;
    for (int i = threadIdx.x; i < 1024; i += 256) { float v = r[i]; s += v * v; }
#pragma unroll
    for (int o = 16; o; o >>= 1) s += __shfl_xor(s, o, 32);
    if ((threadIdx.x & 31) == 0) red[threadIdx.x >> 5] = s;
    __syncthreads();
    if (threadIdx.x == 0) {
        float t = 0.0f;
#pragma unroll
        for (int j = 0; j < 8; ++j) t += red[j];
        invs = 1.0f / fmaxf(sqrtf(t), 1e-12f);
    }
    __syncthreads();
    const float iv = invs;
    for (int i = threadIdx.x; i < 1024; i += 256) out[row * 1024 + i] = r[i] * iv;
}

// ---------------------------------------------------------------------------
// Words epilogue: per-(b,l) L2 norm over FEAT, then (l,f) -> (f,l) transpose.
// One block per batch. Phase 1: wave-reduced norms. Phase 2: 32x32 LDS tiles.
// ---------------------------------------------------------------------------
__global__ __launch_bounds__(256) void words_out_k(const float* __restrict__ codeB,
                                                   float* __restrict__ out) {
    __shared__ float inv[126];
    __shared__ float tile[32][33];
    const int b = blockIdx.x;
    const int t = threadIdx.x;
    const int lane = t & 31;
    const int w = t >> 5;

    for (int r = w; r < 126; r += 8) {
        const float* row = codeB + (long)(b * 126 + r) * 1024;
        float s = 0.0f;
        for (int j = lane; j < 1024; j += 32) { float v = row[j]; s += v * v; }
#pragma unroll
        for (int o = 16; o; o >>= 1) s += __shfl_xor(s, o, 32);
        if (lane == 0) inv[r] = 1.0f / fmaxf(sqrtf(s), 1e-12f);
    }
    __syncthreads();

    float* ob = out + (long)b * 1024 * 126;
    for (int lt = 0; lt < 4; ++lt) {
        for (int ft = 0; ft < 32; ++ft) {
            const int lloc = t >> 3;
            const int f4   = (t & 7) * 4;
            const int l    = lt * 32 + lloc;
            if (l < 126) {
                const float* src = codeB + (long)(b * 126 + l) * 1024 + ft * 32 + f4;
                const float iv = inv[l];
                float4 v = *(const float4*)src;
                tile[f4 + 0][lloc] = v.x * iv;
                tile[f4 + 1][lloc] = v.y * iv;
                tile[f4 + 2][lloc] = v.z * iv;
                tile[f4 + 3][lloc] = v.w * iv;
            }
            __syncthreads();
            const int floc = t >> 3;
            const int l4   = (t & 7) * 4;
            float* dst = ob + (ft * 32 + floc) * 126 + lt * 32;
#pragma unroll
            for (int j = 0; j < 4; ++j) {
                const int l2 = lt * 32 + l4 + j;
                if (l2 < 126) dst[l4 + j] = tile[floc][l4 + j];
            }
            __syncthreads();
        }
    }
}

// ---------------------------------------------------------------------------
// Host launch
// ---------------------------------------------------------------------------
extern "C" void kernel_launch(void* const* d_in, const int* in_sizes, int n_in,
                              void* d_out, int out_size, void* d_ws, size_t ws_size,
                              hipStream_t stream) {
    (void)in_sizes; (void)n_in; (void)out_size; (void)ws_size;
    const float* words = (const float*)d_in[0];
    const float* sentE = (const float*)d_in[1];
    const float* W2 = (const float*)d_in[2]; const float* b2 = (const float*)d_in[3];
    const float* W3 = (const float*)d_in[4]; const float* b3 = (const float*)d_in[5];
    const float* W4 = (const float*)d_in[6]; const float* b4 = (const float*)d_in[7];
    const float* Wp = (const float*)d_in[8]; const float* bp = (const float*)d_in[9];

    char* ws = (char*)d_ws;
    size_t off = 0;
    auto carve = [&](size_t bytes) -> char* {
        char* p = ws + off;
        off += (bytes + 255) & ~(size_t)255;
        return p;
    };
    // ~109 MB total workspace
    bf16* wordsBF = (bf16*)carve(sizeof(bf16) * 128 * 132 * 768);
    bf16* w2bf    = (bf16*)carve(sizeof(bf16) * 1024 * 1536);
    bf16* w3bf    = (bf16*)carve(sizeof(bf16) * 1024 * 2304);
    bf16* w4bf    = (bf16*)carve(sizeof(bf16) * 1024 * 3072);
    bf16* wpbf    = (bf16*)carve(sizeof(bf16) * 1024 * 768);
    bf16* sentBF  = (bf16*)carve(sizeof(bf16) * 128 * 768);
    float* codeB  = (float*)carve(sizeof(float) * 16128 * 1024);
    float* sentRaw= (float*)carve(sizeof(float) * 128 * 1024);

    const int NW = 128 * 132 * 768;
    cvt_words<<<(NW + 255) / 256, 256, 0, stream>>>(words, wordsBF, NW);
    cvt_flat<<<(1024 * 1536 + 255) / 256, 256, 0, stream>>>(W2, w2bf, 1024 * 1536);
    cvt_flat<<<(1024 * 2304 + 255) / 256, 256, 0, stream>>>(W3, w3bf, 1024 * 2304);
    cvt_flat<<<(1024 * 3072 + 255) / 256, 256, 0, stream>>>(W4, w4bf, 1024 * 3072);
    cvt_flat<<<(1024 * 768 + 255) / 256, 256, 0, stream>>>(Wp, wpbf, 1024 * 768);
    cvt_flat<<<(128 * 768 + 255) / 256, 256, 0, stream>>>(sentE, sentBF, 128 * 768);

    // fused 3-branch conv GEMM: M = 128*126, N = 1024
    wmma_gemm<3, true><<<dim3(8, 126), 256, 0, stream>>>(
        wordsBF, 126, 132 * 768,
        w2bf, 1536, b2, 126,
        w3bf, 2304, b3, 125,
        w4bf, 3072, b4, 124,
        codeB);

    // sentence GEMM: M = 128, N = 1024, K = 768 (no relu, no masking)
    wmma_gemm<1, false><<<dim3(8, 1), 256, 0, stream>>>(
        sentBF, 128, 128 * 768,
        wpbf, 768, bp, 1 << 30,
        wpbf, 768, bp, 0,
        wpbf, 768, bp, 0,
        sentRaw);

    float* outWords = (float*)d_out;                     // (128, 1024, 126)
    float* outSent  = (float*)d_out + 128 * 1024 * 126;  // (128, 1024)
    sent_norm<<<128, 256, 0, stream>>>(sentRaw, outSent);
    words_out_k<<<128, 256, 0, stream>>>(codeB, outWords);
}
